// GAVIBSentimentTokenModel_20023137534206
// MI455X (gfx1250) — matrix operand order repaired and checked by reading, not compiled, not run on hardware
//
#include <hip/hip_runtime.h>
#include <hip/hip_bf16.h>
#include <math.h>

typedef _Float16 half_t;
typedef __attribute__((ext_vector_type(16))) _Float16 v16h;
typedef __attribute__((ext_vector_type(8)))  float    v8f;
typedef __attribute__((ext_vector_type(2)))  _Float16 h2;
typedef __attribute__((ext_vector_type(4)))  _Float16 h4;

#define B_  128
#define T_  2048
#define D_  192
#define H_  64
#define BT  (B_ * T_)

#define ROWS_PER_WG 64
#define THREADS_S1  128   // 4 waves of 32; each wave owns a 16-row tile

#define LOGPI    (-1.6094379124341003f)   // log(0.2)
#define LOG1MPI  (-0.22314355131420976f)  // log(0.8)

// ---------------------------------------------------------------------------
// Stage 1: logits[b*T+t] = w2 . relu(x @ w1 + b1) + b2 via WMMA f16->f32.
// Block: 64 token rows staged through LDS (f32->f16), w1 transposed in LDS.
// ---------------------------------------------------------------------------
__global__ __launch_bounds__(THREADS_S1)
void s1_logits(const float* __restrict__ x, const float* __restrict__ w1,
               const float* __restrict__ b1, const float* __restrict__ w2,
               const float* __restrict__ b2, float* __restrict__ logits)
{
    __shared__ half_t lds_a[ROWS_PER_WG * D_];   // [row][k]  24576 B
    __shared__ half_t lds_w1[H_ * D_];           // [col][k]  24576 B
    __shared__ float  lds_b1[H_];
    __shared__ float  lds_w2[H_];

    const int tid     = threadIdx.x;
    const int rowbase = blockIdx.x * ROWS_PER_WG;

    // Coalesced 128B streaming of the token tile, converted to f16 in LDS.
    const float4* gt = (const float4*)(x + (size_t)rowbase * D_);
    for (int i = tid; i < (ROWS_PER_WG * D_) / 4; i += THREADS_S1) {
        float4 v = gt[i];
        h4 h; h[0] = (half_t)v.x; h[1] = (half_t)v.y;
              h[2] = (half_t)v.z; h[3] = (half_t)v.w;
        *(h4*)(&lds_a[4 * i]) = h;
    }
    // w1 is [D][H] row-major in global; store as [col][K] so B-fragment K-pairs
    // are contiguous (one ds_load_b32 per VGPR pair).
    for (int i = tid; i < D_ * H_; i += THREADS_S1) {
        int k = i >> 6, col = i & 63;
        lds_w1[col * D_ + k] = (half_t)w1[i];
    }
    if (tid < H_) { lds_b1[tid] = b1[tid]; lds_w2[tid] = w2[tid]; }
    __syncthreads();

    const int lane = tid & 31;
    const int wave = tid >> 5;
    const int l16  = lane & 15;
    const int hi   = lane >> 4;           // half-wave select

    const half_t* arow  = &lds_a[(wave * 16 + l16) * D_];
    const int     koffA = 8 * hi;         // A layout: K = e + 8*((e>=8)+(lane>=16))
    const int     koffB = 8 * hi;         // B layout: K = e + 8*((e>=8)+(lane>=16))

    v8f acc[4] = {};                      // 4 N-tiles of 16 columns

    #pragma unroll
    for (int kb = 0; kb < D_ / 32; ++kb) {
        const int kbase = kb * 32;
        v16h a;
        #pragma unroll
        for (int p = 0; p < 8; ++p) {     // element pair p -> VGPR p
            int K = kbase + 2 * p + 8 * (p >> 2) + koffA;
            h2 t = *(const h2*)(arow + K);
            a[2 * p] = t[0]; a[2 * p + 1] = t[1];
        }
        #pragma unroll
        for (int nb = 0; nb < 4; ++nb) {
            const half_t* bcol = &lds_w1[(nb * 16 + l16) * D_ + kbase + koffB];
            v16h bf;
            #pragma unroll
            for (int p = 0; p < 8; ++p) {
                h2 t = *(const h2*)(bcol + 2 * p + 8 * (p >> 2));
                bf[2 * p] = t[0]; bf[2 * p + 1] = t[1];
            }
            acc[nb] = __builtin_amdgcn_wmma_f32_16x16x32_f16(
                false, a, false, bf, (short)0, acc[nb], false, false);
        }
    }

    // Epilogue: h = relu(acc + b1); partial = h * w2; reduce over 64 columns.
    // C/D layout: lane n holds N=n%16; VGPR r -> M=r (lanes 0-15), M=r+8 (16-31).
    float s[8];
    #pragma unroll
    for (int r = 0; r < 8; ++r) s[r] = 0.f;
    #pragma unroll
    for (int nb = 0; nb < 4; ++nb) {
        float b1v = lds_b1[nb * 16 + l16];
        float w2v = lds_w2[nb * 16 + l16];
        #pragma unroll
        for (int r = 0; r < 8; ++r) {
            float h = acc[nb][r] + b1v;
            h = h > 0.f ? h : 0.f;
            s[r] += h * w2v;
        }
    }
    // Sum across the 16 lanes of each half-wave (xor masks keep bit 4 fixed).
    #pragma unroll
    for (int r = 0; r < 8; ++r) {
        s[r] += __shfl_xor(s[r], 1, 32);
        s[r] += __shfl_xor(s[r], 2, 32);
        s[r] += __shfl_xor(s[r], 4, 32);
        s[r] += __shfl_xor(s[r], 8, 32);
    }
    if (l16 == 0) {
        float b2v = b2[0];
        int row0 = rowbase + wave * 16 + 8 * hi;
        #pragma unroll
        for (int r = 0; r < 8; ++r) logits[row0 + r] = s[r] + b2v;
    }
}

// ---------------------------------------------------------------------------
// Stage 2: per batch row — z_train, z_soft (LDS), KL row mean, exact top-k
// ranks with JAX's stable-argsort tie behavior.
// ---------------------------------------------------------------------------
__device__ __forceinline__ float logsig(float x) {
    return (x < 0.f) ? (x - log1pf(__expf(x))) : (-log1pf(__expf(-x)));
}

__global__ __launch_bounds__(256)
void s2_row(const float* __restrict__ logits, const float* __restrict__ u,
            const int* __restrict__ am, float* __restrict__ z_train,
            float* __restrict__ z_hard, float* __restrict__ kl_row)
{
    __shared__ float zrow[T_];
    __shared__ float fred[256];
    __shared__ int   ired[256];
    __shared__ int   len_sh;

    const int    b    = blockIdx.x;
    const int    tid  = threadIdx.x;
    const size_t base = (size_t)b * T_;

    int   lcnt = 0;
    float kls  = 0.f;
    #pragma unroll
    for (int j = 0; j < T_ / 256; ++j) {
        int   t  = tid + j * 256;                 // coalesced
        float lg = logits[base + t];
        int   m  = am[base + t];
        float p  = 1.f / (1.f + __expf(-lg));
        zrow[t]  = (m > 0) ? p : 0.f;
        float uu = u[base + t];
        float ln = __logf(uu) - log1pf(-uu);      // logistic noise
        float zt = (m > 0) ? 1.f / (1.f + __expf(-(lg + ln))) : 0.f;  // TAU=1
        z_train[base + t] = zt;
        if (m > 0) {
            ++lcnt;
            float lp  = logsig(lg);
            float l1p = logsig(-lg);
            kls += p * (lp - LOGPI) + (1.f - p) * (l1p - LOG1MPI);
        }
    }
    fred[tid] = kls; ired[tid] = lcnt;
    __syncthreads();
    for (int off = 128; off > 0; off >>= 1) {
        if (tid < off) { fred[tid] += fred[tid + off]; ired[tid] += ired[tid + off]; }
        __syncthreads();
    }
    if (tid == 0) {
        len_sh    = ired[0];
        kl_row[b] = fred[0] / (float)ired[0];
    }
    __syncthreads();

    const int k = (int)ceilf((float)len_sh * 0.2f);   // PI = 0.2
    for (int j = 0; j < T_ / 256; ++j) {
        int   t = tid + j * 256;
        float v = zrow[t];
        int rank = 0;
        for (int q = 0; q < T_; ++q) {
            float zq = zrow[q];
            rank += (zq > v) ? 1 : 0;
            rank += (zq == v && q < t) ? 1 : 0;   // stable-sort tie break
        }
        z_hard[base + t] = (rank < k) ? 1.f : 0.f;
    }
}

// ---------------------------------------------------------------------------
// Stage 3: mean of per-row KL means -> scalar.
// ---------------------------------------------------------------------------
__global__ __launch_bounds__(128)
void s3_mean(const float* __restrict__ kl_row, float* __restrict__ out)
{
    __shared__ float red[128];
    int tid = threadIdx.x;
    red[tid] = kl_row[tid];
    __syncthreads();
    for (int off = 64; off > 0; off >>= 1) {
        if (tid < off) red[tid] += red[tid + off];
        __syncthreads();
    }
    if (tid == 0) out[0] = red[0] / (float)B_;
}

// ---------------------------------------------------------------------------
extern "C" void kernel_launch(void* const* d_in, const int* in_sizes, int n_in,
                              void* d_out, int out_size, void* d_ws, size_t ws_size,
                              hipStream_t stream)
{
    const float* x   = (const float*)d_in[0];   // token_reps [B,T,D]
    const float* w1  = (const float*)d_in[1];   // [D,H]
    const float* b1  = (const float*)d_in[2];   // [H]
    const float* w2  = (const float*)d_in[3];   // [H]
    const float* b2  = (const float*)d_in[4];   // [1]
    const float* u   = (const float*)d_in[5];   // [B,T]
    const int*   am  = (const int*)d_in[6];     // [B,T]

    float* out     = (float*)d_out;
    float* z_train = out;                       // [B,T]
    float* z_hard  = out + BT;                  // [B,T]
    float* kl_out  = out + 2 * BT;              // [1]

    float* logits = (float*)d_ws;               // BT floats
    float* kl_row = logits + BT;                // B_ floats

    s1_logits<<<BT / ROWS_PER_WG, THREADS_S1, 0, stream>>>(x, w1, b1, w2, b2, logits);
    s2_row<<<B_, 256, 0, stream>>>(logits, u, am, z_train, z_hard, kl_row);
    s3_mean<<<1, 128, 0, stream>>>(kl_row, kl_out);
}
